// MultiHeadAttn_39608188404227
// MI455X (gfx1250) — matrix operand order repaired and checked by reading, not compiled, not run on hardware
//
#include <hip/hip_runtime.h>

// ---------------------------------------------------------------------------
// Transformer-XL relative multi-head attention for MI455X (gfx1250).
// All contractions use v_wmma_f32_16x16x32_bf16 (bf16 in, f32 accumulate).
// Attention is fused flash-style (online softmax), so the 2 x 256MB score
// tensors are never materialized; total workspace traffic ~90MB (fits in L2).
// Linear GEMMs stage their shared A tile into LDS via the gfx1250 async
// global->LDS engine (ASYNCcnt) when the toolchain exposes it.
// ---------------------------------------------------------------------------

#define QL   1024
#define NB   4
#define DIN  1024
#define NH   16
#define DH   64
#define MROWS (QL*NB)        // 4096 flattened (q,b) rows
#define E3   (3*NH*DH)       // 3072
#define RKROWS (QL+16)       // Rk padded so rel-shift band never reads OOB
#define SCALE 0.125f         // 1/sqrt(64)
#define ALD  (DIN + 8)       // LDS A-tile row stride (2064B: bank-conflict free)

typedef __bf16 bf16;
typedef __bf16 v8bf  __attribute__((ext_vector_type(8)));
typedef __bf16 v16bf __attribute__((ext_vector_type(16)));
typedef float  v8f   __attribute__((ext_vector_type(8)));

// The ROCm clang-22 async-LDS builtin takes pointers to GCC-style int4
// vectors: AS1 (global, prints as __device__) source, AS3 (LDS) destination.
typedef int v4i __attribute__((vector_size(16)));
typedef __attribute__((address_space(1))) v4i gv4i;   // global int4
typedef __attribute__((address_space(3))) v4i lv4i;   // LDS int4

#if defined(__has_builtin)
#if __has_builtin(__builtin_amdgcn_global_load_async_to_lds_b128)
#define ASYNC_LDS 1
#else
#define ASYNC_LDS 0
#endif
#else
#define ASYNC_LDS 0
#endif

__device__ __forceinline__ v16bf cat8(v8bf a, v8bf b) {
  return __builtin_shufflevector(a, b, 0,1,2,3,4,5,6,7,8,9,10,11,12,13,14,15);
}

// A fragment 16x32 bf16: lane row = lane%16, hi = lane/16.
// elements 0..7  = K (hi*8 .. hi*8+7), elements 8..15 = K (16+hi*8 ..)
__device__ __forceinline__ v16bf load_a32(const bf16* rowp, int k0, int hi) {
  const bf16* p = rowp + k0 + hi*8;
  return cat8(*(const v8bf*)p, *(const v8bf*)(p + 16));
}
// B fragment 32x16 bf16: lane col = lane%16; elements 0..7 = K (hi*8 ..), 8..15 = K (16+hi*8 ..)
__device__ __forceinline__ v16bf load_b32(const bf16* colp, int k0, int hi) {
  const bf16* p = colp + k0 + hi*8;
  return cat8(*(const v8bf*)p, *(const v8bf*)(p + 16));
}

__device__ __forceinline__ v8f wmma_bf16(v16bf a, v16bf b, v8f c) {
  return __builtin_amdgcn_wmma_f32_16x16x32_bf16(false, a, false, b,
                                                 (short)0, c, false, false);
}

// Wave-local LDS producer->consumer sync (waves in a block run divergent
// causal extents, so no block barrier; s_wait_dscnt is the CDNA5 split wait).
__device__ __forceinline__ void wave_sync_lds() {
  __builtin_amdgcn_wave_barrier();
  asm volatile("s_wait_dscnt 0" ::: "memory");
  __builtin_amdgcn_wave_barrier();
}

// 16-byte global->LDS copy: async DMA path on gfx1250 toolchains that expose
// it (tracked by ASYNCcnt), else register round-trip (tracked by DScnt).
__device__ __forceinline__ void copy16_to_lds(bf16* dst, const bf16* src) {
#if ASYNC_LDS
  __builtin_amdgcn_global_load_async_to_lds_b128((gv4i*)src, (lv4i*)dst, 0, 0);
#else
  *(v8bf*)dst = *(const v8bf*)src;
#endif
}

__device__ __forceinline__ void copy_join() {
#if ASYNC_LDS
#if __has_builtin(__builtin_amdgcn_s_wait_asynccnt)
  __builtin_amdgcn_s_wait_asynccnt(0);
#else
  asm volatile("s_wait_asynccnt 0" ::: "memory");
#endif
#endif
  // fallback path: __syncthreads() below covers DScnt via barrier
}

// ---------------------------------------------------------------------------
__global__ __launch_bounds__(256) void k_f32_to_bf16(const float* __restrict__ in,
                                                     bf16* __restrict__ out, int n) {
  int i = blockIdx.x * blockDim.x + threadIdx.x;
  if (i < n) out[i] = (bf16)in[i];
}

__global__ __launch_bounds__(256) void k_zero_bf16(bf16* __restrict__ out, int n) {
  int i = blockIdx.x * blockDim.x + threadIdx.x;
  if (i < n) out[i] = (bf16)0.0f;
}

// ---------------------------------------------------------------------------
// Shared GEMM core: stage the workgroup's 16x1024 A tile into LDS once
// (all 4 waves share it), then K-loop with A from LDS and B from global.
// ---------------------------------------------------------------------------
__device__ __forceinline__ v8f gemm_tile_stageA(
    const bf16* __restrict__ A, const bf16* __restrict__ W,
    bf16 (*s_a)[ALD], int mt, int nt, int n, int hi) {
  const int t = threadIdx.x;
  // 16 rows x 128 chunks of 16B; 16 chunks per thread.
#pragma unroll
  for (int c = t; c < 16*128; c += 128) {
    const int row = c >> 7, chunk = c & 127;
    copy16_to_lds(&s_a[row][chunk*8],
                  A + (size_t)(mt*16 + row)*DIN + chunk*8);
  }
  copy_join();
  __syncthreads();

  const bf16* arow = &s_a[n][0];
  const bf16* wrow = W + (size_t)(nt*16 + n) * DIN;
  v8f acc = {};
  for (int k0 = 0; k0 < DIN; k0 += 32)
    acc = wmma_bf16(load_a32(arow, k0, hi), load_b32(wrow, k0, hi), acc);
  return acc;
}

// ---------------------------------------------------------------------------
// QKV projection: C[4096][3072] = W_in(bf16) x qkv_w(bf16)^T, fused epilogue
// scatters into per-head Qc (= q + r_w_b), Qp (= q + r_r_b), Kc, and V^T.
// ---------------------------------------------------------------------------
__global__ __launch_bounds__(128) void k_gemm_qkv(
    const bf16* __restrict__ A, const bf16* __restrict__ W,
    const float* __restrict__ rwb, const float* __restrict__ rrb,
    bf16* __restrict__ Qc, bf16* __restrict__ Qp,
    bf16* __restrict__ Kc, bf16* __restrict__ Vt) {
  __shared__ alignas(16) bf16 s_a[16][ALD];
  const int wv = threadIdx.x >> 5, lane = threadIdx.x & 31;
  const int n = lane & 15, hi = lane >> 4;
  const int mt = blockIdx.x & 255;
  const int nt = (blockIdx.x >> 8) * 4 + wv;

  v8f acc = gemm_tile_stageA(A, W, s_a, mt, nt, n, hi);

  const int col = nt*16 + n;
  const int sec = col >> 10, cc = col & 1023, h = cc >> 6, d = cc & 63;
#pragma unroll
  for (int v = 0; v < 8; ++v) {
    int row = mt*16 + v + 8*hi;
    int q = row >> 2, b = row & 3;
    float val = acc[v];
    if (sec == 0) {
      size_t off = ((size_t)(b*NH + h)*QL + q)*DH + d;
      Qc[off] = (bf16)(val + rwb[cc]);
      Qp[off] = (bf16)(val + rrb[cc]);
    } else if (sec == 1) {
      Kc[((size_t)(b*NH + h)*QL + q)*DH + d] = (bf16)val;
    } else {
      Vt[((size_t)(b*NH + h)*DH + d)*QL + q] = (bf16)val;   // transposed V
    }
  }
}

// r_k projection into padded Rk[b][h][1040][64] (pad rows pre-zeroed).
__global__ __launch_bounds__(128) void k_gemm_rk(
    const bf16* __restrict__ A, const bf16* __restrict__ W,
    bf16* __restrict__ Rk) {
  __shared__ alignas(16) bf16 s_a[16][ALD];
  const int wv = threadIdx.x >> 5, lane = threadIdx.x & 31;
  const int n = lane & 15, hi = lane >> 4;
  const int mt = blockIdx.x & 255;
  const int nt = (blockIdx.x >> 8) * 4 + wv;

  v8f acc = gemm_tile_stageA(A, W, s_a, mt, nt, n, hi);

  const int col = nt*16 + n, h = col >> 6, d = col & 63;
#pragma unroll
  for (int v = 0; v < 8; ++v) {
    int row = mt*16 + v + 8*hi;
    int q = row >> 2, b = row & 3;
    Rk[((size_t)(b*NH + h)*RKROWS + q)*DH + d] = (bf16)acc[v];
  }
}

// ---------------------------------------------------------------------------
// Fused attention: per wave = one 16-row i-tile of one (b,h).
// rel_shift folded as bd[i,j] = bd_raw[i, j + L-1-i] for j<=i (rest masked).
// ---------------------------------------------------------------------------
__global__ __launch_bounds__(128) void k_attn(
    const bf16* __restrict__ Qc, const bf16* __restrict__ Qp,
    const bf16* __restrict__ Kc, const bf16* __restrict__ Rk,
    const bf16* __restrict__ Vt, bf16* __restrict__ attv) {
  __shared__ float s_band[4][16][36];              // bd_raw 16x32 band (+pad)
  __shared__ float s_sp[4][16][34];                // masked/scaled scores 16x32
  __shared__ alignas(16) bf16 s_pt[4][16][40];     // probs bf16 (A-frag source)
  __shared__ float s_m[4][16], s_l[4][16], s_f[4][16];

  const int wv = threadIdx.x >> 5, lane = threadIdx.x & 31;
  const int n = lane & 15, hi = lane >> 4;

  const int bid = blockIdx.x;
  const int itg = bid & 15;
  const int h   = (bid >> 4) & 15;
  const int bb  = bid >> 8;
  const int it  = itg*4 + wv;          // this wave's i-tile
  const int i0  = it*16;

  const bf16* Qcp = Qc + (size_t)(bb*NH + h) * QL * DH;
  const bf16* Qpp = Qp + (size_t)(bb*NH + h) * QL * DH;
  const bf16* Kcp = Kc + (size_t)(bb*NH + h) * QL * DH;
  const bf16* Rkp = Rk + (size_t)(bb*NH + h) * RKROWS * DH;
  const bf16* Vtp = Vt + (size_t)(bb*NH + h) * DH * QL;

  // Q fragments for this row block, K=64 -> two k-steps, resident in VGPRs.
  const bf16* qr  = Qcp + (size_t)(i0 + n) * DH;
  const bf16* qrp = Qpp + (size_t)(i0 + n) * DH;
  v16bf aQc0 = load_a32(qr, 0, hi),  aQc1 = load_a32(qr, 32, hi);
  v16bf aQp0 = load_a32(qrp, 0, hi), aQp1 = load_a32(qrp, 32, hi);

  v8f O0 = {}, O1 = {}, O2 = {}, O3 = {};
  if (lane < 16) { s_m[wv][lane] = -1e30f; s_l[wv][lane] = 0.0f; }
  wave_sync_lds();

  const int jtlast = it;                   // causal: j-tile <= i-tile
  const int npair  = (jtlast >> 1) + 1;    // 32-column steps
  for (int jp = 0; jp < npair; ++jp) {
    for (int sub = 0; sub < 2; ++sub) {
      const int jt = jp*2 + sub;
      if (jt > jtlast) {                   // odd tail: mask second half
        if (lane < 16)
          for (int c = 0; c < 16; ++c) s_sp[wv][lane][16 + c] = -1e30f;
        wave_sync_lds();
        break;
      }
      const int j0 = jt*16;
      // content scores: (q + r_w_b) . k
      const bf16* kcol = Kcp + (size_t)(j0 + n) * DH;
      v8f ac = {};
      ac = wmma_bf16(aQc0, load_b32(kcol, 0, hi), ac);
      ac = wmma_bf16(aQc1, load_b32(kcol, 32, hi), ac);
      // positional band: bd_raw columns [base, base+32)
      const int base = j0 + QL - 16 - i0;  // >= 0; base+31 <= 1039 < RKROWS
      const bf16* r0 = Rkp + (size_t)(base + n) * DH;
      const bf16* r1 = Rkp + (size_t)(base + 16 + n) * DH;
      v8f bd0 = {}, bd1 = {};
      bd0 = wmma_bf16(aQp0, load_b32(r0, 0, hi), bd0);
      bd0 = wmma_bf16(aQp1, load_b32(r0, 32, hi), bd0);
      bd1 = wmma_bf16(aQp0, load_b32(r1, 0, hi), bd1);
      bd1 = wmma_bf16(aQp1, load_b32(r1, 32, hi), bd1);
#pragma unroll
      for (int v = 0; v < 8; ++v) {
        int m = v + 8*hi;
        s_band[wv][m][n]      = bd0[v];
        s_band[wv][m][16 + n] = bd1[v];
      }
      wave_sync_lds();
#pragma unroll
      for (int v = 0; v < 8; ++v) {
        int m = v + 8*hi;
        float s = (ac[v] + s_band[wv][m][n + 15 - m]) * SCALE; // rel-shift
        if (j0 + n > i0 + m) s = -1e30f;                       // causal mask
        s_sp[wv][m][sub*16 + n] = s;
      }
      wave_sync_lds();
    }
    // ---- online softmax over this 32-column step (one lane per row) ----
    if (lane < 16) {
      const int r = lane;
      float mx = s_m[wv][r], tmax = -1e30f;
      for (int c = 0; c < 32; ++c) tmax = fmaxf(tmax, s_sp[wv][r][c]);
      float mnew = fmaxf(mx, tmax);
      float f = __expf(mx - mnew);
      float psum = 0.0f;
      for (int c = 0; c < 32; ++c) {
        float pv = __expf(s_sp[wv][r][c] - mnew);
        psum += pv;
        s_pt[wv][r][c] = (bf16)pv;
      }
      s_l[wv][r] = s_l[wv][r]*f + psum;
      s_m[wv][r] = mnew;
      s_f[wv][r] = f;
    }
    wave_sync_lds();
    float fs[8];
#pragma unroll
    for (int v = 0; v < 8; ++v) fs[v] = s_f[wv][v + 8*hi];
#pragma unroll
    for (int v = 0; v < 8; ++v) { O0[v]*=fs[v]; O1[v]*=fs[v]; O2[v]*=fs[v]; O3[v]*=fs[v]; }
    // ---- P x V over K=32 (A from LDS probs, B from pre-transposed V) ----
    const bf16* prow = &s_pt[wv][n][0];
    v16bf aP = cat8(*(const v8bf*)(prow + hi*8), *(const v8bf*)(prow + 16 + hi*8));
    const int kb = jp*32;
    const bf16* vcol = Vtp + (size_t)n * QL + kb;
    O0 = wmma_bf16(aP, load_b32(vcol,          0, hi), O0);
    O1 = wmma_bf16(aP, load_b32(vcol + 16*QL,  0, hi), O1);
    O2 = wmma_bf16(aP, load_b32(vcol + 32*QL,  0, hi), O2);
    O3 = wmma_bf16(aP, load_b32(vcol + 48*QL,  0, hi), O3);
    wave_sync_lds();   // protect s_pt/s_sp before next step overwrites
  }
  // ---- epilogue: normalize by running sum, emit bf16 att_v rows ----
  float rl[8];
#pragma unroll
  for (int v = 0; v < 8; ++v) rl[v] = 1.0f / s_l[wv][v + 8*hi];
#pragma unroll
  for (int v = 0; v < 8; ++v) {
    int m = v + 8*hi;
    size_t rowoff = ((size_t)(i0 + m)*NB + bb) * (NH*DH) + (size_t)h*DH;
    attv[rowoff +      n] = (bf16)(O0[v]*rl[v]);
    attv[rowoff + 16 + n] = (bf16)(O1[v]*rl[v]);
    attv[rowoff + 32 + n] = (bf16)(O2[v]*rl[v]);
    attv[rowoff + 48 + n] = (bf16)(O3[v]*rl[v]);
  }
}

// Output projection: att_o[4096][1024] = att_v(bf16) x o_w(bf16)^T  (f32 out)
__global__ __launch_bounds__(128) void k_gemm_out(
    const bf16* __restrict__ A, const bf16* __restrict__ W,
    float* __restrict__ atto) {
  __shared__ alignas(16) bf16 s_a[16][ALD];
  const int wv = threadIdx.x >> 5, lane = threadIdx.x & 31;
  const int n = lane & 15, hi = lane >> 4;
  const int mt = blockIdx.x & 255;
  const int nt = (blockIdx.x >> 8) * 4 + wv;

  v8f acc = gemm_tile_stageA(A, W, s_a, mt, nt, n, hi);

  const int col = nt*16 + n;
#pragma unroll
  for (int v = 0; v < 8; ++v) {
    int row = mt*16 + v + 8*hi;
    atto[(size_t)row*DIN + col] = acc[v];
  }
}

// Residual + LayerNorm, one row (q,b) per workgroup.
__global__ __launch_bounds__(256) void k_ln(
    const float* __restrict__ w, const float* __restrict__ atto,
    const float* __restrict__ g, const float* __restrict__ bta,
    float* __restrict__ out) {
  __shared__ float ss[256], sq[256];
  const int r = blockIdx.x, t = threadIdx.x;
  float x[4], s = 0.0f, q2 = 0.0f;
#pragma unroll
  for (int i = 0; i < 4; ++i) {
    int c = t + i*256;
    float v = w[(size_t)r*DIN + c] + atto[(size_t)r*DIN + c];
    x[i] = v; s += v; q2 += v*v;
  }
  ss[t] = s; sq[t] = q2; __syncthreads();
  for (int st = 128; st > 0; st >>= 1) {
    if (t < st) { ss[t] += ss[t+st]; sq[t] += sq[t+st]; }
    __syncthreads();
  }
  const float mu   = ss[0] * (1.0f/DIN);
  const float var  = sq[0] * (1.0f/DIN) - mu*mu;
  const float rstd = rsqrtf(var + 1e-5f);
#pragma unroll
  for (int i = 0; i < 4; ++i) {
    int c = t + i*256;
    out[(size_t)r*DIN + c] = (x[i] - mu)*rstd*g[c] + bta[c];
  }
}

// ---------------------------------------------------------------------------
extern "C" void kernel_launch(void* const* d_in, const int* in_sizes, int n_in,
                              void* d_out, int out_size, void* d_ws, size_t ws_size,
                              hipStream_t stream) {
  (void)in_sizes; (void)n_in; (void)out_size; (void)ws_size;
  const float* w    = (const float*)d_in[0];
  const float* r    = (const float*)d_in[1];
  const float* rwb  = (const float*)d_in[2];
  const float* rrb  = (const float*)d_in[3];
  // d_in[4] = att_msk (causal triu) -- folded analytically into k_attn
  const float* qkvw = (const float*)d_in[5];
  const float* rww  = (const float*)d_in[6];
  const float* oww  = (const float*)d_in[7];
  const float* lng  = (const float*)d_in[8];
  const float* lnb  = (const float*)d_in[9];

  char* p = (char*)d_ws;
  auto carve = [&](size_t bytes) -> char* {
    char* q = p; p += (bytes + 255) & ~(size_t)255; return q;
  };
  bf16* w_bf    = (bf16*)carve((size_t)MROWS*DIN*2);
  bf16* r_bf    = (bf16*)carve((size_t)MROWS*DIN*2);
  bf16* qkvw_bf = (bf16*)carve((size_t)E3*DIN*2);
  bf16* rww_bf  = (bf16*)carve((size_t)DIN*DIN*2);
  bf16* oww_bf  = (bf16*)carve((size_t)DIN*DIN*2);
  bf16* Qc      = (bf16*)carve((size_t)NB*NH*QL*DH*2);
  bf16* Qp      = (bf16*)carve((size_t)NB*NH*QL*DH*2);
  bf16* Kc      = (bf16*)carve((size_t)NB*NH*QL*DH*2);
  bf16* Vt      = (bf16*)carve((size_t)NB*NH*DH*QL*2);
  bf16* Rk      = (bf16*)carve((size_t)NB*NH*RKROWS*DH*2);
  bf16* attv    = (bf16*)carve((size_t)MROWS*DIN*2);
  float* atto   = (float*)carve((size_t)MROWS*DIN*4);

  auto cvt = [&](const float* in, bf16* out, int nElem) {
    k_f32_to_bf16<<<(nElem + 255)/256, 256, 0, stream>>>(in, out, nElem);
  };
  cvt(w,    w_bf,    MROWS*DIN);
  cvt(r,    r_bf,    MROWS*DIN);
  cvt(qkvw, qkvw_bf, E3*DIN);
  cvt(rww,  rww_bf,  DIN*DIN);
  cvt(oww,  oww_bf,  DIN*DIN);

  const int rkElems = NB*NH*RKROWS*DH;
  k_zero_bf16<<<(rkElems + 255)/256, 256, 0, stream>>>(Rk, rkElems);

  k_gemm_qkv<<<256*48, 128, 0, stream>>>(w_bf, qkvw_bf, rwb, rrb, Qc, Qp, Kc, Vt);
  k_gemm_rk <<<256*16, 128, 0, stream>>>(r_bf, rww_bf, Rk);
  k_attn    <<<NB*NH*16, 128, 0, stream>>>(Qc, Qp, Kc, Rk, Vt, attv);
  k_gemm_out<<<256*16, 128, 0, stream>>>(attv, oww_bf, atto);
  k_ln      <<<MROWS, 256, 0, stream>>>(w, atto, lng, lnb, (float*)d_out);
}
